// DWM_19834158973319
// MI455X (gfx1250) — matrix operand order repaired and checked by reading, not compiled, not run on hardware
//
#include <hip/hip_runtime.h>
#include <stdint.h>

// ---------------- configuration (mirrors the reference) ----------------
#define BATCH      64
#define SEQ        256
#define INDIM      264     // CONTROL_BITS + DATA_BITS
#define STATEU     512
#define HEADS      4
#define MSLOT      32
#define NADDR      256
#define COMBD      904     // 264 + 512 + 128
#define KPAD       928     // 29 * 32
#define KTILES     29
#define NOUT       1192    // 512 (Ws) + 256 (Wo) + 424 (Wu)
#define NPAD       1200
#define NTILES     75
#define UPDSZ      424
#define HP         106
#define BGRP       16      // batch rows per workgroup
#define NWG        4
#define NTHREADS   512     // 16 waves

typedef __attribute__((ext_vector_type(8)))  float   v8f;
typedef __attribute__((ext_vector_type(16))) __bf16  v16bf;
typedef __attribute__((ext_vector_type(8)))  __bf16  v8bf;

// workspace layout
#define WFRAG_DW   ((size_t)NTILES * KTILES * 32 * 8)          // 556800 dwords
#define WFRAG_BYTES (WFRAG_DW * 4)                              // 2,227,200
#define MEM_BYTES  ((size_t)BATCH * NADDR * MSLOT * 4)          // 2,097,152
#define BIAS_BYTES ((size_t)NPAD * 4)
#define SMEM_BYTES 313344

// ---------------- device helpers ----------------
__device__ __forceinline__ float sigf(float x) { return 1.0f / (1.0f + __expf(-x)); }
__device__ __forceinline__ float splusf(float x) {
    return fmaxf(x, 0.0f) + __logf(1.0f + __expf(-fabsf(x)));
}
__device__ __forceinline__ float wsum(float v) {
#pragma unroll
    for (int o = 16; o > 0; o >>= 1) v += __shfl_xor(v, o, 32);
    return v;
}
__device__ __forceinline__ float wmaxr(float v) {
#pragma unroll
    for (int o = 16; o > 0; o >>= 1) v = fmaxf(v, __shfl_xor(v, o, 32));
    return v;
}
__device__ __forceinline__ uint32_t bf16bits(float f) {
    __bf16 b = (__bf16)f;
    return (uint32_t)__builtin_bit_cast(unsigned short, b);
}

// fused weight element [k][n]; zero padded outside COMBD x NOUT
__device__ __forceinline__ float fetchW(int k, int n,
                                        const float* __restrict__ Ws,
                                        const float* __restrict__ Wo,
                                        const float* __restrict__ Wu) {
    if (k >= COMBD || n >= NOUT) return 0.0f;
    if (n < STATEU)      return Ws[(size_t)k * STATEU + n];
    if (n < STATEU + 256) return Wo[(size_t)k * 256 + (n - STATEU)];
    return Wu[(size_t)k * UPDSZ + (n - STATEU - 256)];
}

// ---------------- kernel 1: pre-swizzle weights into WMMA B-fragment order ----------------
// Fragment (nt, kt): 32 lanes x 8 dwords.  Lane L: column n = nt*16 + (L&15),
// half = L>>4 selects K subrange; dword j packs bf16 pair K = kt*32 + half*16 + 2j (+1).
__global__ void dwm_convert(const float* __restrict__ Wo, const float* __restrict__ bo,
                            const float* __restrict__ Ws, const float* __restrict__ bs,
                            const float* __restrict__ Wu, const float* __restrict__ bu,
                            uint32_t* __restrict__ wfrag, float* __restrict__ biascat) {
    int blk = blockIdx.x;
    if (blk < NTILES * KTILES) {
        int nt = blk / KTILES, kt = blk % KTILES;
        int tid = threadIdx.x;          // 256 threads = 32 lanes x 8 dwords
        int lane = tid >> 3, j = tid & 7;
        int hlf = lane >> 4, c16 = lane & 15;
        int n = nt * 16 + c16;
        int klo = kt * 32 + hlf * 16 + 2 * j;
        float lo = fetchW(klo,     n, Ws, Wo, Wu);
        float hi = fetchW(klo + 1, n, Ws, Wo, Wu);
        wfrag[((size_t)blk * 32 + lane) * 8 + j] = bf16bits(lo) | (bf16bits(hi) << 16);
    } else {
        for (int c = threadIdx.x; c < NPAD; c += 256) {
            float v = 0.0f;
            if (c < STATEU)           v = bs[c];
            else if (c < STATEU + 256) v = bo[c - STATEU];
            else if (c < NOUT)         v = bu[c - STATEU - 256];
            biascat[c] = v;
        }
    }
}

// ---------------- kernel 2: the full recurrent cell, 4 WGs x 16 batch rows ----------------
__global__ __launch_bounds__(NTHREADS, 1)
void dwm_recurrent(const float* __restrict__ xin,        // [64][256][264]
                   const uint32_t* __restrict__ wfrag,   // pre-swizzled bf16 B fragments
                   const float* __restrict__ biascat,    // [1200]
                   float* __restrict__ memg,             // [64][256][32]  (b, n, m)
                   float* __restrict__ outg) {           // [64][256][256]
    extern __shared__ char smemraw[];
    float* state = (float*)smemraw;                 // 16*512
    float* wt    = state + BGRP * STATEU;           // 16*4*256
    float* wtd   = wt    + BGRP * 1024;             // 16*4*256
    float* wts   = wtd   + BGRP * 1024;             // 16*4*256 scratch (scores / wt_g)
    float* updb  = wts   + BGRP * 1024;             // 16*424
    float* er    = updb  + BGRP * UPDSZ;            // 16*4*32
    float* ad    = er    + BGRP * 128;              // 16*4*32
    float* kh    = ad    + BGRP * 128;              // 16*4*32
    float* par   = kh    + BGRP * 128;              // 16*4*10  (s0,s1,s2,j0,j1,j2,jd,gamma,beta,g)
    __bf16* comb = (__bf16*)(par + BGRP * 40);      // 16*928 bf16

    const int tid  = threadIdx.x;
    const int wave = tid >> 5, lane = tid & 31;
    const int hlf  = lane >> 4, c16 = lane & 15;
    const int b0   = blockIdx.x * BGRP;
    float* mem = memg + (size_t)b0 * NADDR * MSLOT;

    // ---- init carry ----
    for (int i = tid; i < BGRP * STATEU; i += NTHREADS) state[i] = 1.0f;
    for (int i = tid; i < BGRP * 1024; i += NTHREADS) {
        float v = ((i & 255) == 0) ? 1.0f : 0.0f;
        wt[i] = v; wtd[i] = v;
    }
    for (int i = tid; i < BGRP * 24; i += NTHREADS)
        comb[(i / 24) * KPAD + COMBD + (i % 24)] = (__bf16)0.0f;
    for (int i = tid; i < BGRP * NADDR * MSLOT; i += NTHREADS) mem[i] = 0.01f;
    asm volatile("s_wait_storecnt 0" ::: "memory");
    __syncthreads();

    for (int t = 0; t < SEQ; ++t) {
        // ---- A: build comb = [x_t | state | read] as bf16 in LDS ----
        for (int i = tid; i < BGRP * INDIM; i += NTHREADS) {
            int b = i / INDIM, c = i % INDIM;
            comb[b * KPAD + c] = (__bf16)xin[((size_t)(b0 + b) * SEQ + t) * INDIM + c];
        }
        for (int i = tid; i < BGRP * STATEU; i += NTHREADS) {
            int b = i >> 9, c = i & 511;
            comb[b * KPAD + INDIM + c] = (__bf16)state[b * STATEU + c];
        }
        {   // read[b][h][m] = sum_n wt[b][h][n] * mem[b][n][m]  (one thread per (b,m))
            int b = tid >> 5, m = tid & 31;
            const float* mb = mem + (size_t)b * NADDR * MSLOT;
            const float* wb = wt + b * 1024;
            float a0 = 0.f, a1 = 0.f, a2 = 0.f, a3 = 0.f;
            for (int n = 0; n < NADDR; ++n) {
                float mv = mb[n * MSLOT + m];
                a0 = fmaf(wb[n], mv, a0);
                a1 = fmaf(wb[256 + n], mv, a1);
                a2 = fmaf(wb[512 + n], mv, a2);
                a3 = fmaf(wb[768 + n], mv, a3);
            }
            __bf16* cb = comb + b * KPAD + INDIM + STATEU;
            cb[m] = (__bf16)a0; cb[32 + m] = (__bf16)a1;
            cb[64 + m] = (__bf16)a2; cb[96 + m] = (__bf16)a3;
        }
        __syncthreads();

        // ---- B: controller GEMM  [16 x 928] x [928 x 1200] via WMMA bf16 ----
        for (int nt = wave; nt < NTILES; nt += 16) {
            v8f acc = {0.f, 0.f, 0.f, 0.f, 0.f, 0.f, 0.f, 0.f};
            const uint32_t* bbase = wfrag + (((size_t)nt * KTILES) * 32 + lane) * 8;
            for (int kt = 0; kt < KTILES; ++kt) {
                // A fragment: lane row M = c16; elems 0..15 -> K = kt*32 + hlf*16 + e
                const __bf16* pa = comb + c16 * KPAD + kt * 32 + hlf * 16;
                v8bf alo = *(const v8bf*)pa;
                v8bf ahi = *(const v8bf*)(pa + 8);
                v16bf av = __builtin_shufflevector(alo, ahi,
                            0, 1, 2, 3, 4, 5, 6, 7, 8, 9, 10, 11, 12, 13, 14, 15);
                v16bf bv = *(const v16bf*)(bbase + (size_t)kt * 32 * 8);
                acc = __builtin_amdgcn_wmma_f32_16x16x32_bf16(
                        false, av, false, bv, (short)0, acc, false, false);
            }
            int col = nt * 16 + c16;
            float bias = biascat[col];
#pragma unroll
            for (int r = 0; r < 8; ++r) {
                int b = r + hlf * 8;                 // C layout: row = r + 8*(lane/16)
                float v = acc[r] + bias;
                if (col < STATEU) {
                    state[b * STATEU + col] = sigf(v);
                } else if (col < STATEU + 256) {
                    outg[((size_t)(b0 + b) * SEQ + t) * 256 + (col - STATEU)] = sigf(v);
                } else if (col < NOUT) {
                    updb[b * UPDSZ + (col - STATEU - 256)] = v;
                }
            }
        }
        __syncthreads();

        // ---- C: per-(b,h) interface-parameter transforms ----
        if (tid < BGRP * HEADS) {
            int b = tid >> 2, h = tid & 3;
            const float* u = updb + b * UPDSZ + h * HP;
            float s0 = splusf(u[0]), s1 = splusf(u[1]), s2 = splusf(u[2]);
            float mx = fmaxf(s0, fmaxf(s1, s2));
            float e0 = __expf(s0 - mx), e1 = __expf(s1 - mx), e2 = __expf(s2 - mx);
            float den = e0 + e1 + e2;
            float jm = fmaxf(u[4], fmaxf(u[5], u[6]));
            float j0 = __expf(u[4] - jm), j1 = __expf(u[5] - jm), j2 = __expf(u[6] - jm);
            float jden = j0 + j1 + j2;
            float* p = par + (b * 4 + h) * 10;
            p[0] = e0 / den; p[1] = e1 / den; p[2] = e2 / den;
            p[3] = j0 / jden; p[4] = j1 / jden; p[5] = j2 / jden;
            p[6] = sigf(u[3]);              // jd
            p[7] = 1.0f + splusf(u[7]);     // gamma
            p[8] = splusf(u[104]);          // beta
            p[9] = sigf(u[105]);            // g
            float* ed = er + (b * 4 + h) * 32;
            float* adp = ad + (b * 4 + h) * 32;
            float* kd = kh + (b * 4 + h) * 32;
            float kn = 0.0f;
            for (int m = 0; m < MSLOT; ++m) {
                float ev = sigf(u[8 + m]);
                float kv = tanhf(u[72 + m]);
                ed[m] = ev; adp[m] = u[40 + m]; kd[m] = kv;
                kn += kv * kv;
            }
            float inv = 1.0f / (sqrtf(kn) + 1e-12f);
            for (int m = 0; m < MSLOT; ++m) kd[m] *= inv;
        }
        __syncthreads();

        // ---- D: memory erase/add + column norms + content scores ----
        for (int cidx = tid; cidx < BGRP * NADDR; cidx += NTHREADS) {
            int b = cidx >> 8, n = cidx & 255;
            float* mc = mem + ((size_t)b * NADDR + n) * MSLOT;
            float w0 = wt[b * 1024 + n],       w1 = wt[b * 1024 + 256 + n];
            float w2 = wt[b * 1024 + 512 + n], w3 = wt[b * 1024 + 768 + n];
            const float* eb = er + b * 128;
            const float* ab = ad + b * 128;
            const float* kb = kh + b * 128;
            float ss = 0.f, d0 = 0.f, d1 = 0.f, d2 = 0.f, d3 = 0.f;
            float4* mc4 = (float4*)mc;
#pragma unroll 4
            for (int q = 0; q < 8; ++q) {
                float4 f = mc4[q];
                float vv[4] = {f.x, f.y, f.z, f.w};
#pragma unroll
                for (int s = 0; s < 4; ++s) {
                    int m = q * 4 + s;
                    float v = vv[s];
                    v *= (1.f - eb[m] * w0) * (1.f - eb[32 + m] * w1) *
                         (1.f - eb[64 + m] * w2) * (1.f - eb[96 + m] * w3);
                    v += ab[m] * w0 + ab[32 + m] * w1 + ab[64 + m] * w2 + ab[96 + m] * w3;
                    vv[s] = v;
                    ss += v * v;
                    d0 = fmaf(kb[m], v, d0);      d1 = fmaf(kb[32 + m], v, d1);
                    d2 = fmaf(kb[64 + m], v, d2); d3 = fmaf(kb[96 + m], v, d3);
                }
                mc4[q] = make_float4(vv[0], vv[1], vv[2], vv[3]);
            }
            float invn = 1.0f / (sqrtf(ss) + 1e-12f);
            const float* p = par + b * 40;
            wts[b * 1024 + n]       = p[8]  * d0 * invn;
            wts[b * 1024 + 256 + n] = p[18] * d1 * invn;
            wts[b * 1024 + 512 + n] = p[28] * d2 * invn;
            wts[b * 1024 + 768 + n] = p[38] * d3 * invn;
        }
        asm volatile("s_wait_storecnt 0" ::: "memory");
        __syncthreads();

        // ---- E: softmax / gate / circular-shift / sharpen / snapshot (wave per row) ----
        for (int row = wave; row < BGRP * HEADS; row += 16) {
            int b = row >> 2, h = row & 3;
            float* sc   = wts + b * 1024 + h * 256;
            float* wrow = wt  + b * 1024 + h * 256;
            float* drow = wtd + b * 1024 + h * 256;
            const float* p = par + (b * 4 + h) * 10;
            float s0 = p[0], s1 = p[1], s2 = p[2];
            float j0 = p[3], j1 = p[4], j2 = p[5];
            float jd = p[6], gamma = p[7], g = p[9];

            float v[8];
            float mx = -3.402823e38f;
#pragma unroll
            for (int c = 0; c < 8; ++c) { v[c] = sc[lane + 32 * c]; mx = fmaxf(mx, v[c]); }
            mx = wmaxr(mx);
            float sum = 0.f;
#pragma unroll
            for (int c = 0; c < 8; ++c) { v[c] = __expf(v[c] - mx); sum += v[c]; }
            sum = wsum(sum);
            float invs = 1.0f / sum;
#pragma unroll
            for (int c = 0; c < 8; ++c) {
                int i = lane + 32 * c;
                float wc = v[c] * invs;
                sc[i] = g * wc + (1.0f - g) * wrow[i];   // wt_g
            }
            asm volatile("s_wait_dscnt 0" ::: "memory"); // wave-local LDS visibility

            float pw[8]; float psum = 0.f;
#pragma unroll
            for (int c = 0; c < 8; ++c) {
                int i = lane + 32 * c;
                float y = s0 * sc[(i + 255) & 255] + s1 * sc[i] + s2 * sc[(i + 1) & 255];
                float z = __expf(gamma * __logf(y + 1e-12f));
                pw[c] = z; psum += z;
            }
            psum = wsum(psum);
            float invp = 1.0f / psum;
#pragma unroll
            for (int c = 0; c < 8; ++c) {
                int i = lane + 32 * c;
                float wsv = pw[c] * invp;
                float dnew = (1.0f - jd) * drow[i] + jd * wsv;
                drow[i] = dnew;
                wrow[i] = j0 * wsv + j1 * dnew + j2 * ((i == 0) ? 1.0f : 0.0f);
            }
        }
        __syncthreads();
    }
}

// ---------------- host-side launch ----------------
extern "C" void kernel_launch(void* const* d_in, const int* in_sizes, int n_in,
                              void* d_out, int out_size, void* d_ws, size_t ws_size,
                              hipStream_t stream) {
    (void)in_sizes; (void)n_in; (void)out_size;
    const float* xin = (const float*)d_in[0];
    const float* Wo  = (const float*)d_in[1];
    const float* bo  = (const float*)d_in[2];
    const float* Ws  = (const float*)d_in[3];
    const float* bs  = (const float*)d_in[4];
    const float* Wu  = (const float*)d_in[5];
    const float* bu  = (const float*)d_in[6];
    float* out = (float*)d_out;

    if (ws_size < WFRAG_BYTES + MEM_BYTES + BIAS_BYTES) return;
    uint8_t* ws = (uint8_t*)d_ws;
    uint32_t* wfrag = (uint32_t*)ws;
    float* memg    = (float*)(ws + WFRAG_BYTES);
    float* biascat = (float*)(ws + WFRAG_BYTES + MEM_BYTES);

    dwm_convert<<<NTILES * KTILES + 1, 256, 0, stream>>>(Wo, bo, Ws, bs, Wu, bu, wfrag, biascat);
    dwm_recurrent<<<NWG, NTHREADS, SMEM_BYTES, stream>>>(xin, wfrag, biascat, memg, out);
}
